// ShiftWindowMHA_70669391889259
// MI455X (gfx1250) — compile-verified, matrix operand order checked by reading, not run on hardware
//
#include <hip/hip_runtime.h>
#include <hip/hip_bf16.h>

// ---------------------------------------------------------------------------
// Swin-V2 shifted-window MHA, fused per-window kernel for gfx1250 (MI455X).
// f16 WMMA (V_WMMA_F32_16X16X32_F16) for all GEMMs, f32 accumulation.
// wave32: 8 waves / 256-thread block, one window (64 tokens) per block.
// ---------------------------------------------------------------------------

typedef __attribute__((ext_vector_type(16))) _Float16 v16h;
typedef __attribute__((ext_vector_type(8)))  _Float16 v8h;
typedef __attribute__((ext_vector_type(8)))  float    v8f;

#define WINSZ   8
#define NTOK    64          // tokens per window
#define DM      256         // model dim
#define NHEADS  8
#define HD      32          // head dim
#define XS      264         // f16 row stride for 64x256 LDS buffers (pad 8)
#define VTS     72          // f16 row stride for v^T (256x64) LDS buffer
#define ATS     72          // f16 row stride for attn (64x64 per head)

union V16u { v16h v; v8h h[2]; };

// Load a 16x32 (f16) WMMA A/B fragment from a row-major f16 matrix.
// Lane layout follows the CDNA5 16-bit fragment striping:
//   lanes 0-15 : row = base+lane,    K chunks [0..7] and [16..23]
//   lanes 16-31: row = base+lane-16, K chunks [8..15] and [24..31]
__device__ __forceinline__ v16h load_frag(const _Float16* base, int row0,
                                          int stride, int k0, int lane) {
  const int r  = row0 + (lane & 15);
  const int c0 = k0 + ((lane >> 4) << 3);
  const _Float16* p = base + (size_t)r * stride + c0;
  V16u u;
  u.h[0] = *(const v8h*)(p);
  u.h[1] = *(const v8h*)(p + 16);
  return u.v;
}

__device__ __forceinline__ v8f wmma16(v16h a, v16h b, v8f c) {
  return __builtin_amdgcn_wmma_f32_16x16x32_f16(
      /*neg_a=*/false, a, /*neg_b=*/false, b,
      /*c_mod=*/(short)0, c, /*reuse_a=*/false, /*reuse_b=*/false);
}

// Reductions across a 16-lane group (C/D fragment: N runs across lanes 0-15
// within each half-wave, so xor masks 1/2/4/8 never cross the half boundary).
__device__ __forceinline__ float red16_sum(float v) {
  v += __shfl_xor(v, 1, 32); v += __shfl_xor(v, 2, 32);
  v += __shfl_xor(v, 4, 32); v += __shfl_xor(v, 8, 32);
  return v;
}
__device__ __forceinline__ float red16_max(float v) {
  v = fmaxf(v, __shfl_xor(v, 1, 32)); v = fmaxf(v, __shfl_xor(v, 2, 32));
  v = fmaxf(v, __shfl_xor(v, 4, 32)); v = fmaxf(v, __shfl_xor(v, 8, 32));
  return v;
}

// shift-mask region id: rows/cols [0,8) -> 0, [8,12) -> 1, [12,16) -> 2
__device__ __forceinline__ int reg3(int p) { return p < 8 ? 0 : (p < 12 ? 1 : 2); }

// ---------------------------------------------------------------------------
// One 16x128 slice of a projection GEMM: dst = x(64x256,f16) @ W^T + bias.
// NORM: per-head (32-col) L2 normalization of the f32 accumulators (cosine
//       attention) before f16 conversion.
// TRANS: store transposed (for v, so attn@v B-fragments are contiguous).
// ---------------------------------------------------------------------------
template<bool NORM, bool TRANS>
__device__ __forceinline__ void proj_gemm(const _Float16* src, const _Float16* w16,
                                          const float* bvec, _Float16* dst,
                                          int m0, int n0, int lane) {
  v8f acc[8];
  #pragma unroll
  for (int j = 0; j < 8; ++j) {
    const float bv = bvec[n0 + j * 16 + (lane & 15)];   // C layout: N = lane
    #pragma unroll
    for (int e = 0; e < 8; ++e) acc[j][e] = bv;
  }
  for (int kk = 0; kk < DM; kk += 32) {
    if (kk + 32 < DM)  // gfx1250 global_prefetch_b8 on next weight K-slice
      __builtin_prefetch(w16 + (size_t)(n0 + (lane & 15)) * DM + kk + 32, 0, 1);
    const v16h a = load_frag(src, m0, XS, kk, lane);
    #pragma unroll
    for (int j = 0; j < 8; ++j) {
      const v16h bf = load_frag(w16, n0 + j * 16, DM, kk, lane);
      acc[j] = wmma16(a, bf, acc[j]);
    }
  }
  if (NORM) {
    #pragma unroll
    for (int j = 0; j < 8; j += 2) {        // head = 32 cols = tile pair
      #pragma unroll
      for (int vg = 0; vg < 8; ++vg) {
        float s = acc[j][vg] * acc[j][vg] + acc[j + 1][vg] * acc[j + 1][vg];
        s = red16_sum(s);
        const float rinv = 1.0f / fmaxf(sqrtf(s), 1e-12f);
        acc[j][vg]     *= rinv;
        acc[j + 1][vg] *= rinv;
      }
    }
  }
  #pragma unroll
  for (int j = 0; j < 8; ++j) {
    const int n = n0 + j * 16 + (lane & 15);
    #pragma unroll
    for (int vg = 0; vg < 8; ++vg) {
      const int m = m0 + vg + ((lane >> 4) << 3);
      if (TRANS) dst[(size_t)n * VTS + m] = (_Float16)acc[j][vg];
      else       dst[(size_t)m * XS  + n] = (_Float16)acc[j][vg];
    }
  }
}

// LDS layout (f16 elements). attn aliases x/q/k after they are consumed.
#define OFF_X   0
#define OFF_Q   (NTOK * XS)
#define OFF_K   (2 * NTOK * XS)
#define OFF_VT  (3 * NTOK * XS)
#define OFF_O   (3 * NTOK * XS + DM * VTS)
#define SMEM_ELTS (4 * NTOK * XS + DM * VTS)

__global__ __launch_bounds__(256)
void swinv2_window_attn_kernel(const float* __restrict__ img,
                               const _Float16* __restrict__ wq16, const float* __restrict__ bq,
                               const _Float16* __restrict__ wk16, const float* __restrict__ bk,
                               const _Float16* __restrict__ wv16, const float* __restrict__ bv,
                               const _Float16* __restrict__ wo16, const float* __restrict__ bo,
                               const float* __restrict__ logit_scale,
                               const float* __restrict__ biastab,
                               float* __restrict__ out) {
  extern __shared__ char smem_raw[];
  _Float16* xb = (_Float16*)smem_raw;
  _Float16* qb = xb + OFF_Q;
  _Float16* kb = xb + OFF_K;
  _Float16* vt = xb + OFF_VT;
  _Float16* ob = xb + OFF_O;
  _Float16* at = xb;                 // attn: 8 heads x 64 x ATS, aliases x/q/k

  const int tid  = threadIdx.x;
  const int lane = tid & 31;
  const int wv_id = tid >> 5;

  const int blk = blockIdx.x;        // Bn = B * 4 window-blocks
  const int b  = blk >> 2;
  const int wi = blk & 3;
  const int wy = wi >> 1, wx = wi & 1;

  // ---- Step 1: roll(+4,+4) + window-partition + f32->f16 into LDS ----
  for (int i = tid; i < NTOK * (DM / 4); i += 256) {
    const int row = i >> 6;          // token 0..63
    const int q4  = i & 63;          // float4 index in row
    const int r = row >> 3, c = row & 7;
    const int sh = (((wy << 3) + r) - 4) & 15;   // rolled[h] = img[h-4]
    const int sw = (((wx << 3) + c) - 4) & 15;
    const float4 f =
        ((const float4*)(img + (((size_t)b * 16 + sh) * 16 + sw) * DM))[q4];
    _Float16* dst = xb + (size_t)row * XS + q4 * 4;
    dst[0] = (_Float16)f.x; dst[1] = (_Float16)f.y;
    dst[2] = (_Float16)f.z; dst[3] = (_Float16)f.w;
  }
  __syncthreads();

  // ---- Step 2: q/k/v projections (WMMA). Each wave: 16 rows x 128 cols ----
  const int m0 = (wv_id >> 1) << 4;       // 0,16,32,48
  const int n0 = (wv_id & 1) << 7;        // 0 or 128
  proj_gemm<true,  false>(xb, wq16, bq, qb, m0, n0, lane);   // q, L2-normed
  proj_gemm<true,  false>(xb, wk16, bk, kb, m0, n0, lane);   // k, L2-normed
  proj_gemm<false, true >(xb, wv16, bv, vt, m0, n0, lane);   // v, transposed
  __syncthreads();

  // ---- Step 3: logits = qn@kn^T * scale + bias, mask, softmax (head/wave) --
  {
    const int h = wv_id;
    const float scale = __expf(fminf(logit_scale[h], 4.6051702f)); // log(100)
    const float* bt = biastab + (size_t)h * NTOK * NTOK;

    v8f lg[16];
    #pragma unroll
    for (int t = 0; t < 16; ++t) {
      #pragma unroll
      for (int e = 0; e < 8; ++e) lg[t][e] = 0.0f;
    }

    v16h kf[4];
    #pragma unroll
    for (int nt = 0; nt < 4; ++nt)
      kf[nt] = load_frag(kb, nt * 16, XS, h * HD, lane);   // B = kn^T
    #pragma unroll
    for (int mt = 0; mt < 4; ++mt) {
      const v16h a = load_frag(qb, mt * 16, XS, h * HD, lane);
      #pragma unroll
      for (int nt = 0; nt < 4; ++nt)
        lg[mt * 4 + nt] = wmma16(a, kf[nt], lg[mt * 4 + nt]);   // K = HD = 32
    }

    // scale + relative-position bias + shift-window mask
    #pragma unroll
    for (int mt = 0; mt < 4; ++mt) {
      #pragma unroll
      for (int nt = 0; nt < 4; ++nt) {
        const int Nk = nt * 16 + (lane & 15);               // key token
        const int rk = reg3((wy << 3) + (Nk >> 3)) * 3 +
                       reg3((wx << 3) + (Nk & 7));
        #pragma unroll
        for (int vg = 0; vg < 8; ++vg) {
          const int Mq = mt * 16 + vg + ((lane >> 4) << 3); // query token
          const int rq = reg3((wy << 3) + (Mq >> 3)) * 3 +
                         reg3((wx << 3) + (Mq & 7));
          const float bvl = bt[(size_t)Mq * NTOK + Nk];
          const float v = lg[mt * 4 + nt][vg] * scale + bvl;
          lg[mt * 4 + nt][vg] = (rq == rk) ? v : -1e30f;
        }
      }
    }

    // row softmax fully in registers (row = fixed vg & lane-half; cols span
    // 16 lanes x 4 n-tiles)
    #pragma unroll
    for (int mt = 0; mt < 4; ++mt) {
      #pragma unroll
      for (int vg = 0; vg < 8; ++vg) {
        float mx = fmaxf(fmaxf(lg[mt*4+0][vg], lg[mt*4+1][vg]),
                         fmaxf(lg[mt*4+2][vg], lg[mt*4+3][vg]));
        mx = red16_max(mx);
        float sm = 0.0f;
        #pragma unroll
        for (int nt = 0; nt < 4; ++nt) {
          const float e = __expf(lg[mt*4+nt][vg] - mx);
          lg[mt*4+nt][vg] = e;
          sm += e;
        }
        sm = red16_sum(sm);
        const float rinv = 1.0f / sm;
        #pragma unroll
        for (int nt = 0; nt < 4; ++nt) lg[mt*4+nt][vg] *= rinv;
      }
    }

    __syncthreads();  // all waves done reading qb/kb before aliased writes

    _Float16* ah = at + (size_t)h * NTOK * ATS;
    #pragma unroll
    for (int mt = 0; mt < 4; ++mt) {
      #pragma unroll
      for (int nt = 0; nt < 4; ++nt) {
        const int Nk = nt * 16 + (lane & 15);
        #pragma unroll
        for (int vg = 0; vg < 8; ++vg) {
          const int Mq = mt * 16 + vg + ((lane >> 4) << 3);
          ah[(size_t)Mq * ATS + Nk] = (_Float16)lg[mt * 4 + nt][vg];
        }
      }
    }
  }

  // ---- Step 4: o = attn @ v (one head per wave, WMMA, K = 64 tokens) ----
  {
    const int h = wv_id;
    const _Float16* ah = at + (size_t)h * NTOK * ATS;
    v8f oacc[8];
    #pragma unroll
    for (int t = 0; t < 8; ++t) {
      #pragma unroll
      for (int e = 0; e < 8; ++e) oacc[t][e] = 0.0f;
    }
    #pragma unroll
    for (int kk = 0; kk < NTOK; kk += 32) {
      v16h vf[2];
      #pragma unroll
      for (int nt2 = 0; nt2 < 2; ++nt2)
        vf[nt2] = load_frag(vt, h * HD + nt2 * 16, VTS, kk, lane); // v^T rows
      #pragma unroll
      for (int mt = 0; mt < 4; ++mt) {
        const v16h a = load_frag(ah, mt * 16, ATS, kk, lane);
        #pragma unroll
        for (int nt2 = 0; nt2 < 2; ++nt2)
          oacc[mt * 2 + nt2] = wmma16(a, vf[nt2], oacc[mt * 2 + nt2]);
      }
    }
    #pragma unroll
    for (int mt = 0; mt < 4; ++mt) {
      #pragma unroll
      for (int nt2 = 0; nt2 < 2; ++nt2) {
        const int n = h * HD + nt2 * 16 + (lane & 15);
        #pragma unroll
        for (int vg = 0; vg < 8; ++vg) {
          const int m = mt * 16 + vg + ((lane >> 4) << 3);
          ob[(size_t)m * XS + n] = (_Float16)oacc[mt * 2 + nt2][vg];
        }
      }
    }
  }
  __syncthreads();

  // ---- Step 5: z = o @ wo^T + bo, reverse roll folded into global store ----
  {
    v8f acc[8];
    #pragma unroll
    for (int j = 0; j < 8; ++j) {
      const float bvl = bo[n0 + j * 16 + (lane & 15)];
      #pragma unroll
      for (int e = 0; e < 8; ++e) acc[j][e] = bvl;
    }
    for (int kk = 0; kk < DM; kk += 32) {
      if (kk + 32 < DM)
        __builtin_prefetch(wo16 + (size_t)(n0 + (lane & 15)) * DM + kk + 32, 0, 1);
      const v16h a = load_frag(ob, m0, XS, kk, lane);
      #pragma unroll
      for (int j = 0; j < 8; ++j) {
        const v16h bf = load_frag(wo16, n0 + j * 16, DM, kk, lane);
        acc[j] = wmma16(a, bf, acc[j]);
      }
    }
    #pragma unroll
    for (int j = 0; j < 8; ++j) {
      const int n = n0 + j * 16 + (lane & 15);
      #pragma unroll
      for (int vg = 0; vg < 8; ++vg) {
        const int m  = m0 + vg + ((lane >> 4) << 3);
        const int oh = (((wy << 3) + (m >> 3)) - 4) & 15;  // out = roll(z,-4)
        const int ow = (((wx << 3) + (m & 7)) - 4) & 15;
        out[(((size_t)b * 16 + oh) * 16 + ow) * DM + n] = acc[j][vg];
      }
    }
  }
}

// ---------------------------------------------------------------------------
// Prep kernel 1: f32 -> f16 weight conversion (row-major [out][in] preserved).
// ---------------------------------------------------------------------------
__global__ __launch_bounds__(256)
void swinv2_prep_weights(const float* __restrict__ wq, const float* __restrict__ wk,
                         const float* __restrict__ wv, const float* __restrict__ wo,
                         _Float16* __restrict__ wq16, _Float16* __restrict__ wk16,
                         _Float16* __restrict__ wv16, _Float16* __restrict__ wo16) {
  const int i = blockIdx.x * 256 + threadIdx.x;
  if (i < DM * DM) {
    wq16[i] = (_Float16)wq[i];
    wk16[i] = (_Float16)wk[i];
    wv16[i] = (_Float16)wv[i];
    wo16[i] = (_Float16)wo[i];
  }
}

// ---------------------------------------------------------------------------
// Prep kernel 2: CPB-MLP relative-position bias table -> [H][64][64] f32
// (16 * sigmoid already applied). One thread per (query n, key m) pair.
// ---------------------------------------------------------------------------
__global__ __launch_bounds__(256)
void swinv2_prep_bias(const float* __restrict__ w1, const float* __restrict__ b1,
                      const float* __restrict__ w2, float* __restrict__ biastab) {
  const int p = blockIdx.x * 256 + threadIdx.x;   // 4096 pairs
  if (p >= NTOK * NTOK) return;
  const int n = p >> 6, m = p & 63;
  const float dy = (float)((n >> 3) - (m >> 3));  // relative coords in [-7,7]
  const float dx = (float)((n & 7) - (m & 7));
  const float sy = dy / 7.0f * 8.0f;
  const float sx = dx / 7.0f * 8.0f;
  const float ty = copysignf(log2f(fabsf(sy) + 1.0f) / 3.0f, sy);
  const float tx = copysignf(log2f(fabsf(sx) + 1.0f) / 3.0f, sx);

  float acc[NHEADS];
  #pragma unroll
  for (int h = 0; h < NHEADS; ++h) acc[h] = 0.0f;
  for (int j = 0; j < 512; ++j) {
    const float h1 = fmaxf(w1[j * 2 + 0] * ty + w1[j * 2 + 1] * tx + b1[j], 0.0f);
    #pragma unroll
    for (int h = 0; h < NHEADS; ++h) acc[h] += h1 * w2[h * 512 + j];
  }
  #pragma unroll
  for (int h = 0; h < NHEADS; ++h)
    biastab[(size_t)h * NTOK * NTOK + p] = 16.0f / (1.0f + __expf(-acc[h]));
}

// ---------------------------------------------------------------------------
extern "C" void kernel_launch(void* const* d_in, const int* in_sizes, int n_in,
                              void* d_out, int out_size, void* d_ws, size_t ws_size,
                              hipStream_t stream) {
  (void)in_sizes; (void)n_in; (void)out_size; (void)ws_size;

  const float* img  = (const float*)d_in[0];
  const float* wq   = (const float*)d_in[1];
  const float* bq   = (const float*)d_in[2];
  const float* wk   = (const float*)d_in[3];
  const float* bk   = (const float*)d_in[4];
  const float* wv   = (const float*)d_in[5];
  const float* bv   = (const float*)d_in[6];
  const float* wo   = (const float*)d_in[7];
  const float* bo   = (const float*)d_in[8];
  const float* ls   = (const float*)d_in[9];
  const float* cw1  = (const float*)d_in[10];
  const float* cb1  = (const float*)d_in[11];
  const float* cw2  = (const float*)d_in[12];
  float* out = (float*)d_out;

  // workspace layout (bytes)
  char* ws = (char*)d_ws;
  _Float16* wq16 = (_Float16*)(ws + 0 * 131072);
  _Float16* wk16 = (_Float16*)(ws + 1 * 131072);
  _Float16* wv16 = (_Float16*)(ws + 2 * 131072);
  _Float16* wo16 = (_Float16*)(ws + 3 * 131072);
  float*    btab = (float*)   (ws + 4 * 131072);   // [8][64][64] f32

  swinv2_prep_weights<<<DM * DM / 256, 256, 0, stream>>>(
      wq, wk, wv, wo, wq16, wk16, wv16, wo16);
  swinv2_prep_bias<<<NTOK * NTOK / 256, 256, 0, stream>>>(cw1, cb1, cw2, btab);

  const size_t smem_bytes = (size_t)SMEM_ELTS * sizeof(_Float16);  // 172032 B
  (void)hipFuncSetAttribute((const void*)swinv2_window_attn_kernel,
                            hipFuncAttributeMaxDynamicSharedMemorySize,
                            (int)smem_bytes);

  const int Bn = 512 * 4;   // B * NW windows
  swinv2_window_attn_kernel<<<Bn, 256, smem_bytes, stream>>>(
      img, wq16, bq, wk16, bk, wv16, bv, wo16, bo, ls, btab, out);
}
